// FlowNetC_87385404604738
// MI455X (gfx1250) — matrix operand order repaired and checked by reading, not compiled, not run on hardware
//
#include <hip/hip_runtime.h>

// ---------- types ----------
typedef __attribute__((ext_vector_type(16))) __bf16       v16bf;
typedef __attribute__((ext_vector_type(8)))  float        v8f;
typedef __attribute__((ext_vector_type(4)))  unsigned int vu4;

union Frag { vu4 u[2]; v16bf v; };

__device__ __forceinline__ unsigned short f2bf(float f) {
    unsigned int u = __float_as_uint(f);
    u += 0x7FFFu + ((u >> 16) & 1u);          // round-to-nearest-even
    return (unsigned short)(u >> 16);
}
__device__ __forceinline__ float bf2f(unsigned short h) {
    return __uint_as_float(((unsigned int)h) << 16);
}
__device__ __forceinline__ int refl(int s, int n) {
    s = s < 0 ? -s : s;
    return s >= n ? 2 * n - 2 - s : s;
}

// ---------- pad+convert f32 NCHW -> bf16 NHWC (reflect), channel-pad Cs->Cd ----------
__global__ void k_pad_img(const float* __restrict__ src, unsigned short* __restrict__ dst,
                          int Bn, int Cs, int Cd, int H, int W, int pad) {
    long i = (long)blockIdx.x * blockDim.x + threadIdx.x;
    int Hp = H + 2 * pad, Wp = W + 2 * pad;
    long n = (long)Bn * Hp * Wp * Cd;
    if (i >= n) return;
    int c = (int)(i % Cd); long t = i / Cd;
    int x = (int)(t % Wp); t /= Wp;
    int y = (int)(t % Hp); int b = (int)(t / Hp);
    float v = 0.f;
    if (c < Cs) {
        int sy = refl(y - pad, H), sx = refl(x - pad, W);
        v = src[(((long)b * Cs + c) * H + sy) * W + sx];
    }
    dst[i] = f2bf(v);
}

// ---------- pad bf16 NHWC(strided src) -> bf16 NHWC dense, reflect or zero ----------
__global__ void k_pad(const unsigned short* __restrict__ src, int sStr, int sOff,
                      unsigned short* __restrict__ dst,
                      int Bn, int H, int W, int C, int pad, int reflect) {
    long i = (long)blockIdx.x * blockDim.x + threadIdx.x;
    int Hp = H + 2 * pad, Wp = W + 2 * pad;
    long n = (long)Bn * Hp * Wp * C;
    if (i >= n) return;
    int c = (int)(i % C); long t = i / C;
    int x = (int)(t % Wp); t /= Wp;
    int y = (int)(t % Hp); int b = (int)(t / Hp);
    int sy = y - pad, sx = x - pad;
    unsigned short v = 0;
    if (reflect) {
        sy = refl(sy, H); sx = refl(sx, W);
        v = src[((long)(b * H + sy) * W + sx) * sStr + sOff + c];
    } else if (sy >= 0 && sy < H && sx >= 0 && sx < W) {
        v = src[((long)(b * H + sy) * W + sx) * sStr + sOff + c];
    }
    dst[i] = v;
}

// ---------- weight pack: OIHW f32 (or IO 1x1) -> bf16 [KH][NK][CoPad][32], K = kw*CiL + c ----------
__global__ void k_pack(const float* __restrict__ w, unsigned short* __restrict__ dst,
                       int Co, int Ci, int KH, int KW, int NK, int CoPad, int CiL, int io) {
    long i = (long)blockIdx.x * blockDim.x + threadIdx.x;
    long n = (long)KH * NK * CoPad * 32;
    if (i >= n) return;
    int kl = (int)(i & 31); long t = i >> 5;
    int nn = (int)(t % CoPad); t /= CoPad;
    int kc = (int)(t % NK);    int kh = (int)(t / NK);
    int k = kc * 32 + kl;
    float v = 0.f;
    if (k < KW * CiL && nn < Co) {
        int kw = k / CiL, c = k % CiL;
        if (c < Ci)
            v = io ? w[(long)c * Co + nn]
                   : w[(((long)nn * Ci + c) * KH + kh) * KW + kw];
    }
    dst[i] = f2bf(v);
}

// ---------- implicit-GEMM conv ----------
// One wave computes 16 pixels x (NB*16) output channels. The A fragment is
// loaded once per K-step and reused across NB WMMAs (register blocking over N).
template<int NB>
__global__ void k_conv(const unsigned short* __restrict__ in, int Hp, int Wp, int Cin,
                       const unsigned short* __restrict__ wp, int KH, int NK, int CoPad,
                       unsigned short* __restrict__ out, int Hout, int Wout, int Cout,
                       int outStride, int outOff, int stride,
                       const float* __restrict__ bias, int relu, int Bn) {
    int lane = threadIdx.x;                 // 0..31, wave32
    int m = lane & 15;
    int hi = lane >> 4;                     // 0/1
    int total = Bn * Hout * Wout;
    int p = blockIdx.x * 16 + m;
    if (p >= total) p = total - 1;
    int hw = Hout * Wout;
    int b = p / hw, rp = p % hw;
    int oy = rp / Wout, ox = rp % Wout;
    long rowBase = ((long)(b * Hp + oy * stride) * Wp + ox * stride) * Cin;
    int kb   = hi * 8;                      // A: lanes>=16 hold K base 8 (+16 for 2nd half)
    int koff = hi * 8;                      // B: lanes>=16 hold K base 8 (+16 for 2nd half)
    int nch0 = (blockIdx.y * NB) * 16 + m;  // first output channel handled by this wave

    v8f acc[NB];
#pragma unroll
    for (int j = 0; j < NB; ++j) acc[j] = (v8f){0.f, 0.f, 0.f, 0.f, 0.f, 0.f, 0.f, 0.f};

    for (int kh = 0; kh < KH; ++kh) {
        const unsigned short* ap0 = in + rowBase + (long)kh * Wp * Cin;
        for (int kc = 0; kc < NK; ++kc) {
            Frag A;
            const unsigned short* ap = ap0 + kc * 32 + kb;
            A.u[0] = *(const vu4*)(ap);
            A.u[1] = *(const vu4*)(ap + 16);
            const unsigned short* bbase =
                wp + (((long)kh * NK + kc) * CoPad + nch0) * 32 + koff;
#pragma unroll
            for (int j = 0; j < NB; ++j) {
                Frag Bf;
                const unsigned short* bp = bbase + (long)j * 16 * 32;
                Bf.u[0] = *(const vu4*)(bp);
                Bf.u[1] = *(const vu4*)(bp + 16);
                acc[j] = __builtin_amdgcn_wmma_f32_16x16x32_bf16(
                    false, A.v, false, Bf.v, (short)0, acc[j], false, false);
            }
        }
    }

#pragma unroll
    for (int j = 0; j < NB; ++j) {
        int nch = nch0 + j * 16;
        if (nch < Cout) {
            float bv = bias ? bias[nch] : 0.f;
            for (int r = 0; r < 8; ++r) {
                int pp = blockIdx.x * 16 + (hi << 3) + r;   // D: M = r + 8*(lane>=16)
                if (pp < total) {
                    float v = acc[j][r] + bv;
                    if (relu) v = fmaxf(v, 0.f);
                    out[(long)pp * outStride + outOff + nch] = f2bf(v);
                }
            }
        }
    }
}

// ---------- correlation: block = one pixel, xb vector in LDS, 441 offsets over threads ----------
__global__ void k_corr(const unsigned short* __restrict__ xaPad,
                       const unsigned short* __restrict__ xb,
                       unsigned short* __restrict__ out,
                       int Bn, int S, int C, int outStride) {
    __shared__ float xv[256];
    int p = blockIdx.x;
    int b = p / (S * S), ij = p % (S * S);
    int ii = ij / S, jj = ij % S;
    int tid = threadIdx.x;
    const unsigned short* xbp = xb + (long)p * C;
    for (int c = tid; c < C; c += blockDim.x) xv[c] = bf2f(xbp[c]);
    __syncthreads();
    int Sp = S + 20;
    for (int off = tid; off < 441; off += blockDim.x) {
        int di = off / 21, dj = off % 21;
        const unsigned short* ar =
            xaPad + ((long)(b * Sp + ii + di) * Sp + (jj + dj)) * C;
        float s = 0.f;
        for (int c = 0; c < C; ++c) s += bf2f(ar[c]) * xv[c];
        out[(long)p * outStride + off] = f2bf(s);
    }
}

// ---------- upconv scatter: 2x spatial, even positions relu(z), odd zero ----------
__global__ void k_upconv(const unsigned short* __restrict__ z,
                         int Bn, int h, int w, int C,
                         unsigned short* __restrict__ dst, int dStr, int dOff) {
    int H = 2 * h, W = 2 * w;
    long i = (long)blockIdx.x * blockDim.x + threadIdx.x;
    long n = (long)Bn * H * W * C;
    if (i >= n) return;
    int c = (int)(i % C); long t = i / C;
    int X = (int)(t % W); t /= W;
    int Y = (int)(t % H); int b = (int)(t / H);
    unsigned short v = 0;
    if (!(X & 1) && !(Y & 1)) {
        float f = bf2f(z[((long)(b * h + (Y >> 1)) * w + (X >> 1)) * C + c]);
        if (f < 0.f) f = 0.f;
        v = f2bf(f);
    }
    dst[((long)((b * H + Y) * W + X)) * dStr + dOff + c] = v;
}

// ---------- bilinear upsample (align_corners=True) bf16 -> strided bf16 dest ----------
__global__ void k_upsample(const unsigned short* __restrict__ src,
                           int Bn, int h, int w, int C,
                           unsigned short* __restrict__ dst, int dStr, int dOff, int f) {
    int H = h * f, W = w * f;
    long i = (long)blockIdx.x * blockDim.x + threadIdx.x;
    long n = (long)Bn * H * W * C;
    if (i >= n) return;
    int c = (int)(i % C); long t = i / C;
    int X = (int)(t % W); t /= W;
    int Y = (int)(t % H); int b = (int)(t / H);
    float sy = (float)Y * (float)(h - 1) / (float)(H - 1);
    float sx = (float)X * (float)(w - 1) / (float)(W - 1);
    int i0 = (int)floorf(sy); if (i0 > h - 2) i0 = h - 2; if (i0 < 0) i0 = 0;
    int j0 = (int)floorf(sx); if (j0 > w - 2) j0 = w - 2; if (j0 < 0) j0 = 0;
    float fy = sy - (float)i0, fx = sx - (float)j0;
    const unsigned short* s = src + ((long)(b * h + i0) * w + j0) * C + c;
    long rw = (long)w * C;
    float v00 = bf2f(s[0]),  v01 = bf2f(s[C]);
    float v10 = bf2f(s[rw]), v11 = bf2f(s[rw + C]);
    float v = (v00 * (1 - fx) + v01 * fx) * (1 - fy) + (v10 * (1 - fx) + v11 * fx) * fy;
    dst[((long)((b * H + Y) * W + X)) * dStr + dOff + c] = f2bf(v);
}

// ---------- final x4 bilinear upsample, bf16 NHWC -> f32 NCHW ----------
__global__ void k_upfinal(const unsigned short* __restrict__ src,
                          int Bn, int h, int w, int C, float* __restrict__ out, int f) {
    int H = h * f, W = w * f;
    long i = (long)blockIdx.x * blockDim.x + threadIdx.x;
    long n = (long)Bn * C * H * W;
    if (i >= n) return;
    int X = (int)(i % W); long t = i / W;
    int Y = (int)(t % H); t /= H;
    int c = (int)(t % C); int b = (int)(t / C);
    float sy = (float)Y * (float)(h - 1) / (float)(H - 1);
    float sx = (float)X * (float)(w - 1) / (float)(W - 1);
    int i0 = (int)floorf(sy); if (i0 > h - 2) i0 = h - 2; if (i0 < 0) i0 = 0;
    int j0 = (int)floorf(sx); if (j0 > w - 2) j0 = w - 2; if (j0 < 0) j0 = 0;
    float fy = sy - (float)i0, fx = sx - (float)j0;
    const unsigned short* s = src + ((long)(b * h + i0) * w + j0) * C + c;
    long rw = (long)w * C;
    float v00 = bf2f(s[0]),  v01 = bf2f(s[C]);
    float v10 = bf2f(s[rw]), v11 = bf2f(s[rw + C]);
    out[i] = (v00 * (1 - fx) + v01 * fx) * (1 - fy) + (v10 * (1 - fx) + v11 * fx) * fy;
}

// =====================================================================================
extern "C" void kernel_launch(void* const* d_in, const int* in_sizes, int n_in,
                              void* d_out, int out_size, void* d_ws, size_t ws_size,
                              hipStream_t stream) {
    (void)n_in; (void)out_size; (void)ws_size;
    const float* im1    = (const float*)d_in[0];
    const float* im2    = (const float*)d_in[1];
    const float* wA1    = (const float*)d_in[2];
    const float* wA2    = (const float*)d_in[3];
    const float* wA3    = (const float*)d_in[4];
    const float* wB1    = (const float*)d_in[5];
    const float* wB2    = (const float*)d_in[6];
    const float* wB3    = (const float*)d_in[7];
    const float* wredir = (const float*)d_in[8];
    const float* bredir = (const float*)d_in[9];
    const float* w31    = (const float*)d_in[10];
    const float* w4     = (const float*)d_in[11];
    const float* w41    = (const float*)d_in[12];
    const float* w5     = (const float*)d_in[13];
    const float* w51    = (const float*)d_in[14];
    const float* w6     = (const float*)d_in[15];
    const float* wd5    = (const float*)d_in[16];
    const float* wd4    = (const float*)d_in[17];
    const float* wd3    = (const float*)d_in[18];
    const float* wd2    = (const float*)d_in[19];
    const float* wf5    = (const float*)d_in[20];
    const float* wf4    = (const float*)d_in[21];
    const float* wf3    = (const float*)d_in[22];
    const float* wpred  = (const float*)d_in[23];
    float* outp = (float*)d_out;

    const int Bn = in_sizes[0] / (3 * 384 * 384);
    const int S2 = 200, S3 = 392, S4 = 776, SC = 480;  // channel-padded concat strides

    char* base = (char*)d_ws; size_t cur = 0;
    auto alloc = [&](long elems) -> unsigned short* {
        unsigned short* pptr = (unsigned short*)(base + cur);
        cur += (size_t)((elems * 2 + 1023) & ~1023L);   // 1KB slack per buffer (A-tail overread)
        return pptr;
    };

    // ---- activations ----
    unsigned short* imPad = alloc((long)Bn * 390 * 390 * 4);
    unsigned short* c1    = alloc((long)Bn * 192 * 192 * 64);
    unsigned short* c1Pad = alloc((long)Bn * 196 * 196 * 64);
    unsigned short* y2    = alloc((long)Bn * 96 * 96 * S2);     // [upconv2 64 | a 128 | upflow3 2]
    unsigned short* bTmp  = alloc((long)Bn * 96 * 96 * 128);
    unsigned short* aPad  = alloc((long)Bn * 100 * 100 * 128);
    unsigned short* corrA = alloc((long)Bn * 48 * 48 * 256);
    unsigned short* corrB = alloc((long)Bn * 48 * 48 * 256);
    unsigned short* xaPad = alloc((long)Bn * 68 * 68 * 256);
    unsigned short* ycorr = alloc((long)Bn * 48 * 48 * SC);     // [corr 441 | redir 32]
    unsigned short* ycPad = alloc((long)Bn * 50 * 50 * SC);
    unsigned short* y3    = alloc((long)Bn * 48 * 48 * S3);     // [upconv3 128 | x2 256 | upflow4 2]
    unsigned short* x2Pad = alloc((long)Bn * 50 * 50 * 256);
    unsigned short* t4    = alloc((long)Bn * 24 * 24 * 512);
    unsigned short* t4Pad = alloc((long)Bn * 26 * 26 * 512);
    unsigned short* y4    = alloc((long)Bn * 24 * 24 * S4);     // [upconv4 256 | x3 512 | upflow5 2]
    unsigned short* x3Pad = alloc((long)Bn * 26 * 26 * 512);
    unsigned short* t5    = alloc((long)Bn * 12 * 12 * 512);
    unsigned short* t5Pad = alloc((long)Bn * 14 * 14 * 512);
    unsigned short* y5    = alloc((long)Bn * 12 * 12 * 1024);   // [upconv5 512 | x4 512]
    unsigned short* x4Pad = alloc((long)Bn * 14 * 14 * 512);
    unsigned short* x5    = alloc((long)Bn * 6 * 6 * 1024);
    unsigned short* z5    = alloc((long)Bn * 36 * 512);
    unsigned short* z4    = alloc((long)Bn * 144 * 256);
    unsigned short* z3    = alloc((long)Bn * 2304 * 128);
    unsigned short* z2    = alloc((long)Bn * 9216 * 64);
    unsigned short* y5Z   = alloc((long)Bn * 16 * 16 * 1024);
    unsigned short* y4Z   = alloc((long)Bn * 28 * 28 * S4);
    unsigned short* y3Z   = alloc((long)Bn * 52 * 52 * S3);
    unsigned short* y2Z   = alloc((long)Bn * 100 * 100 * S2);
    unsigned short* f5    = alloc((long)Bn * 144 * 2);
    unsigned short* f4    = alloc((long)Bn * 576 * 2);
    unsigned short* f3    = alloc((long)Bn * 2304 * 2);
    unsigned short* pr    = alloc((long)Bn * 9216 * 2);

    // ---- weight packing ----
    auto packw = [&](const float* wsrc, int Co, int Ci, int KH, int KW,
                     int CoPad, int CiL, int io, int& NK) -> unsigned short* {
        NK = (KW * CiL + 31) / 32;
        long n = (long)KH * NK * CoPad * 32;
        unsigned short* dst = alloc(n);
        k_pack<<<(unsigned)((n + 255) / 256), 256, 0, stream>>>(
            wsrc, dst, Co, Ci, KH, KW, NK, CoPad, CiL, io);
        return dst;
    };
    int nkA1, nkA2, nkA3, nkB1, nkB2, nkB3, nkRd, nk31, nk4, nk41, nk5, nk51, nk6;
    int nkD5, nkD4, nkD3, nkD2, nkF5, nkF4, nkF3, nkPr;
    unsigned short* pA1 = packw(wA1, 64, 3, 7, 7, 64, 4, 0, nkA1);
    unsigned short* pA2 = packw(wA2, 128, 64, 5, 5, 128, 64, 0, nkA2);
    unsigned short* pA3 = packw(wA3, 256, 128, 5, 5, 256, 128, 0, nkA3);
    unsigned short* pB1 = packw(wB1, 64, 3, 7, 7, 64, 4, 0, nkB1);
    unsigned short* pB2 = packw(wB2, 128, 64, 5, 5, 128, 64, 0, nkB2);
    unsigned short* pB3 = packw(wB3, 256, 128, 5, 5, 256, 128, 0, nkB3);
    unsigned short* pRd = packw(wredir, 32, 256, 1, 1, 32, 256, 0, nkRd);
    unsigned short* p31 = packw(w31, 256, 473, 3, 3, 256, SC, 0, nk31);
    unsigned short* p4  = packw(w4,  512, 256, 3, 3, 512, 256, 0, nk4);
    unsigned short* p41 = packw(w41, 512, 512, 3, 3, 512, 512, 0, nk41);
    unsigned short* p5  = packw(w5,  512, 512, 3, 3, 512, 512, 0, nk5);
    unsigned short* p51 = packw(w51, 512, 512, 3, 3, 512, 512, 0, nk51);
    unsigned short* p6  = packw(w6, 1024, 512, 3, 3, 1024, 512, 0, nk6);
    unsigned short* pD5 = packw(wd5, 512, 1024, 1, 1, 512, 1024, 1, nkD5);
    unsigned short* pD4 = packw(wd4, 256, 1024, 1, 1, 256, 1024, 1, nkD4);
    unsigned short* pD3 = packw(wd3, 128, 770, 1, 1, 128, S4, 1, nkD3);
    unsigned short* pD2 = packw(wd2, 64, 386, 1, 1, 64, S3, 1, nkD2);
    unsigned short* pF5 = packw(wf5, 2, 1024, 5, 5, 16, 1024, 0, nkF5);
    unsigned short* pF4 = packw(wf4, 2, 770, 5, 5, 16, S4, 0, nkF4);
    unsigned short* pF3 = packw(wf3, 2, 386, 5, 5, 16, S3, 0, nkF3);
    unsigned short* pPr = packw(wpred, 2, 194, 5, 5, 16, S2, 0, nkPr);

    // ---- helpers ----
    auto conv = [&](const unsigned short* in, int Hp, int Wp, int Cin,
                    const unsigned short* wp, int KH, int NK, int CoPad,
                    unsigned short* o, int Ho, int Wo, int Co, int oStr, int oOff,
                    int st, const float* bias, int relu) {
        int total = Bn * Ho * Wo;
        unsigned gx = (unsigned)((total + 15) / 16);
        if (CoPad % 64 == 0) {
            dim3 g(gx, (unsigned)(CoPad / 64));
            k_conv<4><<<g, dim3(32), 0, stream>>>(in, Hp, Wp, Cin, wp, KH, NK, CoPad,
                                                  o, Ho, Wo, Co, oStr, oOff, st, bias, relu, Bn);
        } else if (CoPad % 32 == 0) {
            dim3 g(gx, (unsigned)(CoPad / 32));
            k_conv<2><<<g, dim3(32), 0, stream>>>(in, Hp, Wp, Cin, wp, KH, NK, CoPad,
                                                  o, Ho, Wo, Co, oStr, oOff, st, bias, relu, Bn);
        } else {
            dim3 g(gx, (unsigned)(CoPad / 16));
            k_conv<1><<<g, dim3(32), 0, stream>>>(in, Hp, Wp, Cin, wp, KH, NK, CoPad,
                                                  o, Ho, Wo, Co, oStr, oOff, st, bias, relu, Bn);
        }
    };
    auto pad = [&](const unsigned short* src, int sStr, int sOff, unsigned short* dst,
                   int Hh, int Ww, int C, int p, int rf) {
        long n = (long)Bn * (Hh + 2 * p) * (Ww + 2 * p) * C;
        k_pad<<<(unsigned)((n + 255) / 256), 256, 0, stream>>>(src, sStr, sOff, dst, Bn, Hh, Ww, C, p, rf);
    };
    auto upconv = [&](const unsigned short* z, int h, int w, int C,
                      unsigned short* dst, int dStr, int dOff) {
        long n = (long)Bn * 4 * h * w * C;
        k_upconv<<<(unsigned)((n + 255) / 256), 256, 0, stream>>>(z, Bn, h, w, C, dst, dStr, dOff);
    };
    auto ups = [&](const unsigned short* src, int h, int w, int C,
                   unsigned short* dst, int dStr, int dOff, int f) {
        long n = (long)Bn * (h * f) * (w * f) * C;
        k_upsample<<<(unsigned)((n + 255) / 256), 256, 0, stream>>>(src, Bn, h, w, C, dst, dStr, dOff, f);
    };
    auto padimg = [&](const float* src) {
        long n = (long)Bn * 390 * 390 * 4;
        k_pad_img<<<(unsigned)((n + 255) / 256), 256, 0, stream>>>(src, imPad, Bn, 3, 4, 384, 384, 3);
    };

    // ---- feature tower, image 1 (keeps 'a' in y2 @ ch 64) ----
    padimg(im1);
    conv(imPad, 390, 390, 4, pA1, 7, nkA1, 64, c1, 192, 192, 64, 64, 0, 2, nullptr, 1);
    pad(c1, 64, 0, c1Pad, 192, 192, 64, 2, 1);
    conv(c1Pad, 196, 196, 64, pA2, 5, nkA2, 128, y2, 96, 96, 128, S2, 64, 2, nullptr, 1);
    pad(y2, S2, 64, aPad, 96, 96, 128, 2, 1);
    conv(aPad, 100, 100, 128, pA3, 5, nkA3, 256, corrA, 48, 48, 256, 256, 0, 2, nullptr, 1);

    // ---- feature tower, image 2 ----
    padimg(im2);
    conv(imPad, 390, 390, 4, pB1, 7, nkB1, 64, c1, 192, 192, 64, 64, 0, 2, nullptr, 1);
    pad(c1, 64, 0, c1Pad, 192, 192, 64, 2, 1);
    conv(c1Pad, 196, 196, 64, pB2, 5, nkB2, 128, bTmp, 96, 96, 128, 128, 0, 2, nullptr, 1);
    pad(bTmp, 128, 0, aPad, 96, 96, 128, 2, 1);
    conv(aPad, 100, 100, 128, pB3, 5, nkB3, 256, corrB, 48, 48, 256, 256, 0, 2, nullptr, 1);

    // ---- correlation + redirect -> ycorr (473 used of 480) ----
    pad(corrA, 256, 0, xaPad, 48, 48, 256, 10, 0);
    k_corr<<<(unsigned)(Bn * 48 * 48), 128, 0, stream>>>(xaPad, corrB, ycorr, Bn, 48, 256, SC);
    conv(corrA, 48, 48, 256, pRd, 1, nkRd, 32, ycorr, 48, 48, 32, SC, 441, 1, bredir, 1);

    // ---- encoder ----
    pad(ycorr, SC, 0, ycPad, 48, 48, SC, 1, 1);
    conv(ycPad, 50, 50, SC, p31, 3, nk31, 256, y3, 48, 48, 256, S3, 128, 1, nullptr, 1);   // x2
    pad(y3, S3, 128, x2Pad, 48, 48, 256, 1, 1);
    conv(x2Pad, 50, 50, 256, p4, 3, nk4, 512, t4, 24, 24, 512, 512, 0, 2, nullptr, 1);
    pad(t4, 512, 0, t4Pad, 24, 24, 512, 1, 1);
    conv(t4Pad, 26, 26, 512, p41, 3, nk41, 512, y4, 24, 24, 512, S4, 256, 1, nullptr, 1);  // x3
    pad(y4, S4, 256, x3Pad, 24, 24, 512, 1, 1);
    conv(x3Pad, 26, 26, 512, p5, 3, nk5, 512, t5, 12, 12, 512, 512, 0, 2, nullptr, 1);
    pad(t5, 512, 0, t5Pad, 12, 12, 512, 1, 1);
    conv(t5Pad, 14, 14, 512, p51, 3, nk51, 512, y5, 12, 12, 512, 1024, 512, 1, nullptr, 1); // x4
    pad(y5, 1024, 512, x4Pad, 12, 12, 512, 1, 1);
    conv(x4Pad, 14, 14, 512, p6, 3, nk6, 1024, x5, 6, 6, 1024, 1024, 0, 2, nullptr, 1);     // x5

    // ---- decoder level 5 ----
    conv(x5, 6, 6, 1024, pD5, 1, nkD5, 512, z5, 6, 6, 512, 512, 0, 1, nullptr, 0);
    upconv(z5, 6, 6, 512, y5, 1024, 0);
    pad(y5, 1024, 0, y5Z, 12, 12, 1024, 2, 0);
    conv(y5Z, 16, 16, 1024, pF5, 5, nkF5, 16, f5, 12, 12, 2, 2, 0, 1, nullptr, 0);          // flow5
    // ---- level 4 ----
    conv(y5, 12, 12, 1024, pD4, 1, nkD4, 256, z4, 12, 12, 256, 256, 0, 1, nullptr, 0);
    upconv(z4, 12, 12, 256, y4, S4, 0);
    ups(f5, 12, 12, 2, y4, S4, 768, 2);
    pad(y4, S4, 0, y4Z, 24, 24, S4, 2, 0);
    conv(y4Z, 28, 28, S4, pF4, 5, nkF4, 16, f4, 24, 24, 2, 2, 0, 1, nullptr, 0);            // flow4
    // ---- level 3 ----
    conv(y4, 24, 24, S4, pD3, 1, nkD3, 128, z3, 24, 24, 128, 128, 0, 1, nullptr, 0);
    upconv(z3, 24, 24, 128, y3, S3, 0);
    ups(f4, 24, 24, 2, y3, S3, 384, 2);
    pad(y3, S3, 0, y3Z, 48, 48, S3, 2, 0);
    conv(y3Z, 52, 52, S3, pF3, 5, nkF3, 16, f3, 48, 48, 2, 2, 0, 1, nullptr, 0);            // flow3
    // ---- level 2 ----
    conv(y3, 48, 48, S3, pD2, 1, nkD2, 64, z2, 48, 48, 64, 64, 0, 1, nullptr, 0);
    upconv(z2, 48, 48, 64, y2, S2, 0);
    ups(f3, 48, 48, 2, y2, S2, 192, 2);
    pad(y2, S2, 0, y2Z, 96, 96, S2, 2, 0);
    conv(y2Z, 100, 100, S2, pPr, 5, nkPr, 16, pr, 96, 96, 2, 2, 0, 1, nullptr, 0);          // pred

    // ---- final x4 bilinear -> f32 NCHW output ----
    {
        long n = (long)Bn * 2 * 384 * 384;
        k_upfinal<<<(unsigned)((n + 255) / 256), 256, 0, stream>>>(pr, Bn, 96, 96, 2, outp, 4);
    }
}
